// PostProcessCocoGrounding_10153302688160
// MI455X (gfx1250) — compile-verified, matrix operand order checked by reading, not run on hardware
//
#include <hip/hip_runtime.h>
#include <hip/hip_bf16.h>
#include <math.h>

// Problem constants (match reference setup_inputs exactly).
constexpr int B = 128, Q = 900, T = 256, L = 92, NSEL = 300;
constexpr int QT = (Q + 15) / 16;   // 57 tiles over Q
constexpr int LT = (L + 15) / 16;   // 6 tiles over L

typedef __attribute__((ext_vector_type(16))) _Float16 v16h;
typedef __attribute__((ext_vector_type(2)))  __fp16   v2fp16;  // cvt_pkrtz result type
typedef __attribute__((ext_vector_type(8)))  float    v8f;

// Fast sigmoid: v_mul + v_exp_f32 + v_add + v_rcp_f32 (4 VALU ops).
// rcp is ~1 ulp; inputs are quantized to f16 for WMMA anyway.
__device__ __forceinline__ float sigmoidf(float x) {
  float e = __builtin_amdgcn_exp2f(x * -1.44269504088896340736f);  // exp(-x)
  return __builtin_amdgcn_rcpf(1.0f + e);
}

// Pack two f32 into one dword of f16 (v_cvt_pk_rtz_f16_f32).
__device__ __forceinline__ unsigned pk2(float x, float y) {
  v2fp16 h = __builtin_amdgcn_cvt_pkrtz(x, y);
  return __builtin_bit_cast(unsigned, h);
}

// ---------------------------------------------------------------------------
// Pass 0: positive_maps f32 -> f16 (B operand; reused by 57 waves/batch, so
// converting once is cheaper than fusing into the GEMM).
// ---------------------------------------------------------------------------
__global__ void cvt_to_f16(const float* __restrict__ in,
                           _Float16* __restrict__ out, int n) {
  int i = blockIdx.x * blockDim.x + threadIdx.x;
  int stride = gridDim.x * blockDim.x;
  for (; i < n; i += stride) out[i] = (_Float16)in[i];
}

// ---------------------------------------------------------------------------
// Pass 1: fused sigmoid + batched GEMM
//   prob[b,q,l] = sum_t sigmoid(logits[b,q,t]) * pm[b,l,t]
// One wave32 computes a 16x92 row strip: 6 accumulators, 48x
// v_wmma_f32_16x16x32_f16 over K=256. A (logits) is loaded as f32 exactly
// once, sigmoid'ed in-register, packed to f16 pairs with v_cvt_pkrtz.
// Grid sized so every wave owns a valid strip -> EXEC all-ones for WMMA.
// ---------------------------------------------------------------------------
__global__ __launch_bounds__(256)
void prob_gemm_wmma(const float*    __restrict__ logits,   // [B,Q,T] f32
                    const _Float16* __restrict__ pmH,      // [B,L,T] f16
                    float* __restrict__ prob) {            // [B,Q,L] f32
  const int lane  = threadIdx.x & 31;
  const int wid   = blockIdx.x * (blockDim.x >> 5) + (threadIdx.x >> 5);
  const int b     = wid / QT;
  const int qt    = wid % QT;

  const int half_ = lane >> 4;       // 0: lanes 0-15, 1: lanes 16-31
  const int mn    = lane & 15;       // row (A) / col (B) within a tile

  // Clamp edge loads; stores are predicated instead.
  const int q  = qt * 16 + mn;
  const int qc = q < Q ? q : Q - 1;

  const float4* rowA =
      (const float4*)(logits + (size_t)b * Q * T + (size_t)qc * T);

  // One f16 row pointer per L-tile (dword-indexed; K pairs are 4B aligned).
  const unsigned* rowB[LT];
#pragma unroll
  for (int lt = 0; lt < LT; ++lt) {
    const int l  = lt * 16 + mn;
    const int lc = l < L ? l : L - 1;
    rowB[lt] = (const unsigned*)(pmH + (size_t)b * L * T + (size_t)lc * T);
  }

  v8f acc[LT];
#pragma unroll
  for (int lt = 0; lt < LT; ++lt) acc[lt] = v8f{};

#pragma unroll
  for (int s = 0; s < 8; ++s) {            // K = 256 in steps of 32
    const int k0 = s * 32 + half_ * 8;     // ISA half-wave K split (halves)

    // A fragment: elements e<4 hold K pairs k0+2e, e>=4 hold k0+16+2(e-4).
    // As f32 that is floats [k0, k0+8) and [k0+16, k0+24): 4x b128 loads.
    const float4 f0 = rowA[(k0 >> 2) + 0];
    const float4 f1 = rowA[(k0 >> 2) + 1];
    const float4 f2 = rowA[(k0 >> 2) + 4];
    const float4 f3 = rowA[(k0 >> 2) + 5];

    union { v16h v; unsigned u[8]; } a;
    a.u[0] = pk2(sigmoidf(f0.x), sigmoidf(f0.y));
    a.u[1] = pk2(sigmoidf(f0.z), sigmoidf(f0.w));
    a.u[2] = pk2(sigmoidf(f1.x), sigmoidf(f1.y));
    a.u[3] = pk2(sigmoidf(f1.z), sigmoidf(f1.w));
    a.u[4] = pk2(sigmoidf(f2.x), sigmoidf(f2.y));
    a.u[5] = pk2(sigmoidf(f2.z), sigmoidf(f2.w));
    a.u[6] = pk2(sigmoidf(f3.x), sigmoidf(f3.y));
    a.u[7] = pk2(sigmoidf(f3.z), sigmoidf(f3.w));

#pragma unroll
    for (int lt = 0; lt < LT; ++lt) {
      union { v16h v; unsigned u[8]; } bf;
#pragma unroll
      for (int e = 0; e < 8; ++e) {
        const int kb = (e < 4) ? (k0 + 2 * e) : (k0 + 16 + 2 * (e - 4));
        bf.u[e] = rowB[lt][kb >> 1];
      }
      acc[lt] = __builtin_amdgcn_wmma_f32_16x16x32_f16(
          /*neg_a=*/false, a.v, /*neg_b=*/false, bf.v,
          /*c_mod=*/(short)0, acc[lt], /*reuse_a=*/false, /*reuse_b=*/false);
    }
  }

  // D layout: N = lane&15, M = vgpr + 8*(lane>>4). Predicated edge stores.
#pragma unroll
  for (int lt = 0; lt < LT; ++lt) {
    const int l = lt * 16 + mn;
    if (l < L) {
      float* col = prob + (size_t)b * Q * L + l;
#pragma unroll
      for (int r = 0; r < 8; ++r) {
        const int m = qt * 16 + r + 8 * half_;
        if (m < Q) col[(size_t)m * L] = acc[lt][r];
      }
    }
  }
}

// ---------------------------------------------------------------------------
// Pass 2: per-batch top-300 (radix threshold on float bits + bitonic sort),
// fused with box cxcywh->xyxy conversion, gather and target-size scaling.
// prob values are strictly positive -> bit pattern order == value order.
// ---------------------------------------------------------------------------
__global__ __launch_bounds__(512)
void topk_postprocess(const float* __restrict__ prob,         // [B,Q,L]
                      const float* __restrict__ pred_boxes,   // [B,Q,4]
                      const int*   __restrict__ target_sizes, // [B,2]
                      float* __restrict__ out) {
  constexpr int NV = Q * L;       // 82800 candidates per batch
  constexpr int NBIN = 2048;      // top 11 bits of positive float
  __shared__ int   hist[NBIN];
  __shared__ float svals[512];
  __shared__ int   sidx[512];
  __shared__ int   s_t, s_above, s_cntA, s_cntT;

  const int b   = blockIdx.x;
  const int tid = threadIdx.x;
  const float* p = prob + (size_t)b * NV;

  for (int i = tid; i < NBIN; i += blockDim.x) hist[i] = 0;
  __syncthreads();
  for (int i = tid; i < NV; i += blockDim.x)
    atomicAdd(&hist[__float_as_uint(p[i]) >> 21], 1);
  __syncthreads();

  if (tid == 0) {
    int cum = 0, t = 0, above = 0;
    for (int bin = NBIN - 1; bin >= 0; --bin) {
      int c = hist[bin];
      if (cum + c >= NSEL) { t = bin; above = cum; break; }
      cum += c;
    }
    s_t = t; s_above = above; s_cntA = 0; s_cntT = 0;
  }
  __syncthreads();
  const int t = s_t, above = s_above;

  for (int i = tid; i < 512; i += blockDim.x) { svals[i] = -INFINITY; sidx[i] = 0; }
  __syncthreads();

  // Compact: everything above threshold bin is in; threshold bin fills rest.
  for (int i = tid; i < NV; i += blockDim.x) {
    float v = p[i];
    int bin = (int)(__float_as_uint(v) >> 21);
    if (bin > t) {
      int slot = atomicAdd(&s_cntA, 1);
      svals[slot] = v; sidx[slot] = i;
    } else if (bin == t) {
      int slot = above + atomicAdd(&s_cntT, 1);
      if (slot < NSEL) { svals[slot] = v; sidx[slot] = i; }
    }
  }
  __syncthreads();

  // Bitonic sort 512 entries, descending by value (carry index).
  for (int k = 2; k <= 512; k <<= 1) {
    for (int j = k >> 1; j > 0; j >>= 1) {
      const int ixj = tid ^ j;
      if (ixj > tid) {
        const bool desc = ((tid & k) == 0);
        float a = svals[tid], c2 = svals[ixj];
        if (desc ? (a < c2) : (a > c2)) {
          svals[tid] = c2; svals[ixj] = a;
          int ti = sidx[tid]; sidx[tid] = sidx[ixj]; sidx[ixj] = ti;
        }
      }
      __syncthreads();
    }
  }

  // Outputs: scores | labels | boxes, concatenated flat in return order.
  if (tid < NSEL) {
    const float v  = svals[tid];
    const int idx  = sidx[tid];
    const int lab  = idx % L;
    const int q    = idx / L;

    out[(size_t)b * NSEL + tid] = v;                                   // scores
    out[(size_t)B * NSEL + (size_t)b * NSEL + tid] = (float)lab;       // labels

    const float* bx = pred_boxes + ((size_t)b * Q + q) * 4;
    const float cx = bx[0], cy = bx[1], w = bx[2], h = bx[3];
    const float ts0 = (float)target_sizes[b * 2 + 0];
    const float ts1 = (float)target_sizes[b * 2 + 1];
    float* ob = out + (size_t)2 * B * NSEL + ((size_t)b * NSEL + tid) * 4;
    ob[0] = (cx - 0.5f * w) * ts1;   // scale_fct = [ts1, ts0, ts1, ts0]
    ob[1] = (cy - 0.5f * h) * ts0;
    ob[2] = (cx + 0.5f * w) * ts1;
    ob[3] = (cy + 0.5f * h) * ts0;
  }
}

// ---------------------------------------------------------------------------
extern "C" void kernel_launch(void* const* d_in, const int* in_sizes, int n_in,
                              void* d_out, int out_size, void* d_ws, size_t ws_size,
                              hipStream_t stream) {
  const float* pred_logits   = (const float*)d_in[0];   // [B,Q,T]
  const float* pred_boxes    = (const float*)d_in[1];   // [B,Q,4]
  const float* positive_maps = (const float*)d_in[2];   // [B,L,T]
  const int*   target_sizes  = (const int*)d_in[3];     // [B,2]
  // d_in[4] = num_select (300) — fixed by the reference setup.

  // Workspace layout: pm f16 (6MB) | prob f32 (42MB)
  _Float16* pmH  = (_Float16*)d_ws;
  float*    prob = (float*)(pmH + (size_t)B * L * T);

  const int nB = B * L * T;   // 3,014,656

  cvt_to_f16<<<2048, 256, 0, stream>>>(positive_maps, pmH, nB);

  // 128 * 57 = 7,296 row-strip waves; 8 waves/block -> exactly 912 blocks.
  const int nblocks = (B * QT) / 8;
  prob_gemm_wmma<<<nblocks, 256, 0, stream>>>(pred_logits, pmH, prob);

  topk_postprocess<<<B, 512, 0, stream>>>(prob, pred_boxes, target_sizes,
                                          (float*)d_out);
}
